// MPCNN_5669356832705
// MI455X (gfx1250) — matrix operand order repaired and checked by reading, not compiled, not run on hardware
//
#include <hip/hip_runtime.h>
#include <math.h>
#include <stdint.h>

typedef _Float16 f16;
typedef __attribute__((ext_vector_type(16))) _Float16 v16h;
typedef __attribute__((ext_vector_type(8)))  _Float16 v8h;
typedef __attribute__((ext_vector_type(8)))  float    v8f;

// Pointee/addrspace types matching __builtin_amdgcn_global_load_async_to_lds_b128:
// param0: int __attribute__((vector_size(16))) __device__(AS1) *
// param1: LDS(AS3) counterpart
typedef int v4i __attribute__((vector_size(16)));
typedef __attribute__((address_space(1))) v4i gv4i_t;
typedef __attribute__((address_space(3))) v4i lv4i_t;

#if __has_builtin(__builtin_amdgcn_global_load_async_to_lds_b128) && \
    __has_builtin(__builtin_amdgcn_s_wait_asynccnt)
#define HAVE_ASYNC_LDS 1
#else
#define HAVE_ASYNC_LDS 0
#endif

#define BATCH 128
#define SEQ   64
#define CDIM  300
#define CPAD  320          // K padding unit (multiple of 32)
#define NHOL  300
#define MPAD  304          // 19 tiles of 16
#define KPF   20
#define HID   150
#define NFEAT 44427
#define KPADF 44448        // 1389 * 32
#define NPAD2 160          // 10 tiles of 16

// ---------------- embedding gather: f16 (padded C=320) + f32 copies ----------------
__global__ void k_gather(const int* __restrict__ sa, const int* __restrict__ sb,
                         const float* __restrict__ emb,
                         f16* __restrict__ sentF16, float* __restrict__ sentF32) {
  int c = threadIdx.x;            // 0..319
  int l = blockIdx.x;             // 0..63
  int b = blockIdx.y;             // 0..127
  int s = blockIdx.z;             // 0..1
  const int* sent = s ? sb : sa;
  int tok = sent[b * SEQ + l];
  float v = 0.f;
  if (c < CDIM) v = emb[(long)tok * CDIM + c];
  sentF16[((long)(s * BATCH + b) * SEQ + l) * CPAD + c] = (f16)v;
  if (c < CDIM)
    sentF32[((long)(s * BATCH + b) * SEQ + l) * CDIM + c] = v;
}

// ---------------- pack holistic conv weights: A[w][MPAD][CPAD*ws] f16, im2col K = t*CPAD + c ----------------
__global__ void k_prepA(const float* __restrict__ hw1, const float* __restrict__ hw2,
                        const float* __restrict__ hw3, f16* __restrict__ A) {
  int idx = blockIdx.x * blockDim.x + threadIdx.x;
  const int sz1 = MPAD * CPAD, sz2 = MPAD * CPAD * 2, sz3 = MPAD * CPAD * 3;
  if (idx >= sz1 + sz2 + sz3) return;
  int w, rem, lda;
  if (idx < sz1)            { w = 0; rem = idx;             lda = CPAD;     }
  else if (idx < sz1 + sz2) { w = 1; rem = idx - sz1;       lda = CPAD * 2; }
  else                      { w = 2; rem = idx - sz1 - sz2; lda = CPAD * 3; }
  int m = rem / lda, kk = rem % lda;
  int t = kk / CPAD, cp = kk % CPAD;
  float v = 0.f;
  if (m < NHOL && cp < CDIM) {
    const float* hw = (w == 0) ? hw1 : (w == 1) ? hw2 : hw3;
    v = hw[(m * CDIM + cp) * (w + 1) + t];   // hw layout [NH][C][ws]
  }
  A[idx] = (f16)v;
}

// ---------------- pack W1 transposed: W1T[NPAD2][KPADF] f16 (N-major so B-frag loads are contiguous) ----------------
__global__ void k_prepW1(const float* __restrict__ W1, f16* __restrict__ W1T) {
  long idx = (long)blockIdx.x * blockDim.x + threadIdx.x;
  long total = (long)NPAD2 * KPADF;
  if (idx >= total) return;
  int n = (int)(idx / KPADF);
  long k = idx % KPADF;
  float v = (n < HID && k < NFEAT) ? W1[k * HID + n] : 0.f;
  W1T[idx] = (f16)v;
}

// ---------------- holistic conv as implicit-im2col WMMA GEMM ----------------
// Block = 8 waves sharing one M-tile. A panel [16][lda] is staged once into LDS
// via GLOBAL_LOAD_ASYNC_TO_LDS_B128 (ASYNCcnt), then all waves read A fragments
// from LDS while im2col B columns stream from global (L2-resident).
__global__ void __launch_bounds__(256)
k_conv_gemm(const f16* __restrict__ A, const f16* __restrict__ sentS,
            float* __restrict__ Cout, int ws, int Lo) {
  const int N = BATCH * Lo;
  const int lda = CPAD * ws;                 // 320 * ws halves per A row
  const int Kiters = (CPAD * ws) >> 5;       // 10 * ws
  __shared__ __attribute__((aligned(32))) f16 Alds[16 * CPAD * 3];  // up to 30 KB

  int tm = blockIdx.y;                       // 0..18 (M tile)
  // ---- stage A panel into LDS (16 * lda halves = 2*lda 16B chunks) ----
  const f16* gpanel = A + (long)tm * 16 * lda;
  for (int i = threadIdx.x; i < 2 * lda; i += 256) {
#if HAVE_ASYNC_LDS
    __builtin_amdgcn_global_load_async_to_lds_b128(
        (gv4i_t*)(uintptr_t)(gpanel + i * 8),
        (lv4i_t*)(unsigned)(uintptr_t)(Alds + i * 8), 0, 0);
#else
    *(v8h*)(Alds + i * 8) = *(const v8h*)(gpanel + i * 8);
#endif
  }
#if HAVE_ASYNC_LDS
  __builtin_amdgcn_s_wait_asynccnt(0);
#endif
  __syncthreads();

  int lane = threadIdx.x & 31;
  int laneLo = lane & 15, laneHi = lane >> 4;
  int tn = blockIdx.x * 8 + (threadIdx.x >> 5);   // N tile per wave (exact grid)

  // A fragment addressing (16-bit A 16x32 layout) -- from LDS
  const f16* aRow = Alds + laneLo * lda + laneHi * 16;
  // B fragment: one im2col column per lane
  int n = tn * 16 + laneLo;
  int b = n / Lo, l = n - b * Lo;
  const f16* bCol = sentS + (long)(b * SEQ + l) * CPAD;
  int kbOfs = laneHi * 16;

  v8f acc = {};
  for (int kt = 0; kt < Kiters; ++kt) {
    int k0 = kt * 32;
    union { v16h v; v8h h[2]; } ua, ub;
    const v8h* pa = (const v8h*)(aRow + k0);   // LDS: ds_load_b128
    ua.h[0] = pa[0];      // K = k0 + sel*16 + 0..7
    ua.h[1] = pa[1];      // K = k0 + sel*16 + 8..15
    int kb = k0 + kbOfs;
    int t = kb / CPAD, cp = kb - t * CPAD;     // stays within one tap (320 % 16 == 0)
    const v8h* pb = (const v8h*)(bCol + (long)t * CPAD + cp);
    ub.h[0] = pb[0];
    ub.h[1] = pb[1];
    __builtin_prefetch(bCol + (long)t * CPAD + cp + 64, 0, 1);
    acc = __builtin_amdgcn_wmma_f32_16x16x32_f16(false, ua.v, false, ub.v,
                                                 (short)0, acc, false, false);
  }
  int col = tn * 16 + laneLo;
  int rbase = tm * 16 + laneHi * 8;
#pragma unroll
  for (int v = 0; v < 8; ++v)
    Cout[(long)(rbase + v) * N + col] = acc[v];
}

// ---------------- tanh + {max, max(-), mean} over Lo for holistic conv ----------------
__global__ void k_pool_h(const float* __restrict__ Cout, const float* __restrict__ hb,
                         float* __restrict__ apool, int sIdx, int w, int Lo) {
  int i = blockIdx.x * blockDim.x + threadIdx.x;
  if (i >= BATCH * NHOL) return;
  int b = i / NHOL, nh = i % NHOL;
  int N = BATCH * Lo;
  float bias = hb[nh];
  const float* row = Cout + (long)nh * N + b * Lo;
  float mx = -1e30f, mn = -1e30f, sm = 0.f;
  for (int l = 0; l < Lo; ++l) {
    float h = tanhf(row[l] + bias);
    mx = fmaxf(mx, h); mn = fmaxf(mn, -h); sm += h;
  }
  long plane = (long)BATCH * NHOL;
  apool[((long)(sIdx * 3 + 0) * 3 + w) * plane + i] = mx;
  apool[((long)(sIdx * 3 + 1) * 3 + w) * plane + i] = mn;
  apool[((long)(sIdx * 3 + 2) * 3 + w) * plane + i] = sm / Lo;
}

// ---------------- inf pools over flat [C*L] ----------------
__global__ void k_pool_inf(const float* __restrict__ sentF32, float* __restrict__ infp) {
  int bi = blockIdx.x;                  // s*BATCH + b
  int s = bi / BATCH, b = bi % BATCH;
  const float* x = sentF32 + (long)(s * BATCH + b) * SEQ * CDIM;
  __shared__ float smx[256], smn[256], ssm[256];
  int tid = threadIdx.x;
  float mx = -1e30f, mn = -1e30f, sm = 0.f;
  for (int i = tid; i < SEQ * CDIM; i += blockDim.x) {
    float v = x[i];
    mx = fmaxf(mx, v); mn = fmaxf(mn, -v); sm += v;
  }
  smx[tid] = mx; smn[tid] = mn; ssm[tid] = sm;
  __syncthreads();
  for (int st = 128; st > 0; st >>= 1) {
    if (tid < st) {
      smx[tid] = fmaxf(smx[tid], smx[tid + st]);
      smn[tid] = fmaxf(smn[tid], smn[tid + st]);
      ssm[tid] += ssm[tid + st];
    }
    __syncthreads();
  }
  if (tid == 0) {
    infp[(s * 3 + 0) * BATCH + b] = smx[0];
    infp[(s * 3 + 1) * BATCH + b] = smn[0];
    infp[(s * 3 + 2) * BATCH + b] = ssm[0] / (float)(SEQ * CDIM);
  }
}

// ---------------- per-dim grouped conv + tanh + max/max(-), stored [s][pool][w][B][K][C] ----------------
__global__ void k_perdim(const float* __restrict__ sentF32,
                         const float* __restrict__ pw1, const float* __restrict__ pb1,
                         const float* __restrict__ pw2, const float* __restrict__ pb2,
                         const float* __restrict__ pw3, const float* __restrict__ pb3,
                         float* __restrict__ bpool) {
  int idx = blockIdx.x * blockDim.x + threadIdx.x;
  const int total = 2 * BATCH * CDIM * KPF;
  if (idx >= total) return;
  int k = idx % KPF; int r = idx / KPF;
  int c = r % CDIM; r /= CDIM;
  int b = r % BATCH; int s = r / BATCH;
  const float* x = sentF32 + (long)(s * BATCH + b) * SEQ * CDIM + c;   // stride CDIM over l
  int ck = c * KPF + k;
  for (int w = 0; w < 3; ++w) {
    int wsv = w + 1, Lo = SEQ - w;
    const float* pw = (w == 0) ? pw1 : (w == 1) ? pw2 : pw3;
    const float* pbp = (w == 0) ? pb1 : (w == 1) ? pb2 : pb3;
    float bias = pbp[ck];
    float wt[3];
    for (int t = 0; t < wsv; ++t) wt[t] = pw[ck * wsv + t];
    float mx = -1e30f, mn = -1e30f;
    for (int l = 0; l < Lo; ++l) {
      float acc = bias;
      for (int t = 0; t < wsv; ++t) acc += wt[t] * x[(l + t) * CDIM];
      float h = tanhf(acc);
      mx = fmaxf(mx, h); mn = fmaxf(mn, -h);
    }
    bpool[((((long)(s * 2 + 0) * 3 + w) * BATCH + b) * KPF + k) * CDIM + c] = mx;
    bpool[((((long)(s * 2 + 1) * 3 + w) * BATCH + b) * KPF + k) * CDIM + c] = mn;
  }
}

// ---------------- feature assembly: exact reference concat order, output f16 [B][KPADF] ----------------
__global__ void __launch_bounds__(256)
k_features(const float* __restrict__ apool, const float* __restrict__ infp,
           const float* __restrict__ bpool, f16* __restrict__ featF16) {
  int b = blockIdx.x, tid = threadIdx.x, bs = blockDim.x;
  __shared__ float s0[256], s1[256], s2[256], s3[256];
  f16* fr = featF16 + (long)b * KPADF;
  int off = 0;

  auto pair = [&](const float* x1, const float* x2, bool adiff) {
    float dot = 0, n1 = 0, n2 = 0, dd = 0;
    for (int c = tid; c < CDIM; c += bs) {
      float a = x1[c], bb = x2[c];
      dot += a * bb; n1 += a * a; n2 += bb * bb;
      float d = a - bb + 1e-6f; dd += d * d;
    }
    s0[tid] = dot; s1[tid] = n1; s2[tid] = n2; s3[tid] = dd;
    __syncthreads();
    for (int st = bs >> 1; st > 0; st >>= 1) {
      if (tid < st) { s0[tid] += s0[tid + st]; s1[tid] += s1[tid + st];
                      s2[tid] += s2[tid + st]; s3[tid] += s3[tid + st]; }
      __syncthreads();
    }
    if (tid == 0) {
      fr[off]     = (f16)(s0[0] / fmaxf(sqrtf(s1[0]) * sqrtf(s2[0]), 1e-8f));
      fr[off + 1] = (f16)sqrtf(s3[0]);
    }
    __syncthreads();
    if (adiff)
      for (int c = tid; c < CDIM; c += bs)
        fr[off + 2 + c] = (f16)fabsf(x1[c] - x2[c]);
    off += adiff ? (2 + CDIM) : 2;
  };
  auto scal = [&](float x1, float x2, bool adiff) {
    if (tid == 0) {
      fr[off]     = (f16)((x1 * x2) / fmaxf(fabsf(x1) * fabsf(x2), 1e-8f));
      fr[off + 1] = (f16)fabsf(x1 - x2 + 1e-6f);
      if (adiff) fr[off + 2] = (f16)fabsf(x1 - x2);
    }
    off += adiff ? 3 : 2;
  };
  const long plane = (long)BATCH * NHOL;
  auto AP = [&](int s, int p, int w) {
    return apool + ((long)(s * 3 + p) * 3 + w) * plane + (long)b * NHOL;
  };

  // horiz: pool-major, ws in [1,2,3,inf]
  for (int p = 0; p < 3; ++p) {
    for (int w = 0; w < 3; ++w) pair(AP(0, p, w), AP(1, p, w), false);
    scal(infp[(0 * 3 + p) * BATCH + b], infp[(1 * 3 + p) * BATCH + b], false);
  }
  // vert-a: 9 finite (ws1,ws2) pairs then inf-inf, per pool
  for (int p = 0; p < 3; ++p) {
    for (int w1 = 0; w1 < 3; ++w1)
      for (int w2 = 0; w2 < 3; ++w2)
        pair(AP(0, p, w1), AP(1, p, w2), true);
    scal(infp[(0 * 3 + p) * BATCH + b], infp[(1 * 3 + p) * BATCH + b], true);
  }
  // vert-b: per-dim blocks, pools (max,min), k-major within (pool,ws)
  for (int p = 0; p < 2; ++p)
    for (int w = 0; w < 3; ++w)
      for (int k = 0; k < KPF; ++k) {
        const float* x1 = bpool + ((((long)(0 * 2 + p) * 3 + w) * BATCH + b) * KPF + k) * CDIM;
        const float* x2 = bpool + ((((long)(1 * 2 + p) * 3 + w) * BATCH + b) * KPF + k) * CDIM;
        pair(x1, x2, true);
      }
  // zero the K padding tail (keeps the MLP WMMA exact)
  for (int j = NFEAT + tid; j < KPADF; j += bs) fr[j] = (f16)0.f;
}

// ---------------- MLP layer 1 as WMMA GEMM: [128][44448] x [44448][160] ----------------
__global__ void __launch_bounds__(256)
k_mlp_gemm(const f16* __restrict__ Afeat, const f16* __restrict__ W1T,
           float* __restrict__ h32) {
  const int ntilesM = BATCH / 16;            // 8 ; tiles = 8*10 = 80 = 10 blocks * 8 waves
  int wid = blockIdx.x * 8 + (threadIdx.x >> 5);
  int lane = threadIdx.x & 31;
  int laneLo = lane & 15, laneHi = lane >> 4;
  int tm = wid % ntilesM;
  int tn = wid / ntilesM;
  const f16* aRow = Afeat + (long)(tm * 16 + laneLo) * KPADF + laneHi * 16;
  const f16* bRow = W1T   + (long)(tn * 16 + laneLo) * KPADF + laneHi * 16;
  v8f acc = {};
  for (int kt = 0; kt < KPADF / 32; ++kt) {
    int k0 = kt * 32;
    union { v16h v; v8h h[2]; } ua, ub;
    const v8h* pa = (const v8h*)(aRow + k0);
    ua.h[0] = pa[0];
    ua.h[1] = pa[1];
    const v8h* pb = (const v8h*)(bRow + k0);
    ub.h[0] = pb[0];
    ub.h[1] = pb[1];
    __builtin_prefetch(aRow + k0 + 128, 0, 1);
    acc = __builtin_amdgcn_wmma_f32_16x16x32_f16(false, ua.v, false, ub.v,
                                                 (short)0, acc, false, false);
  }
  int col = tn * 16 + laneLo;
  int rbase = tm * 16 + laneHi * 8;
#pragma unroll
  for (int v = 0; v < 8; ++v)
    h32[(long)(rbase + v) * NPAD2 + col] = acc[v];
}

// ---------------- head: tanh, W2, log_softmax ----------------
__global__ void k_head(const float* __restrict__ h32, const float* __restrict__ b1,
                       const float* __restrict__ W2, const float* __restrict__ b2,
                       float* __restrict__ out) {
  int b = threadIdx.x;
  if (b >= BATCH) return;
  float l0 = b2[0], l1 = b2[1];
  for (int j = 0; j < HID; ++j) {
    float h = tanhf(h32[b * NPAD2 + j] + b1[j]);
    l0 += h * W2[j * 2 + 0];
    l1 += h * W2[j * 2 + 1];
  }
  float m = fmaxf(l0, l1);
  float lse = m + logf(expf(l0 - m) + expf(l1 - m));
  out[b * 2 + 0] = l0 - lse;
  out[b * 2 + 1] = l1 - lse;
}

extern "C" void kernel_launch(void* const* d_in, const int* in_sizes, int n_in,
                              void* d_out, int out_size, void* d_ws, size_t ws_size,
                              hipStream_t stream) {
  (void)in_sizes; (void)n_in; (void)out_size; (void)ws_size;
  const int*   sa  = (const int*)d_in[0];
  const int*   sb  = (const int*)d_in[1];
  const float* emb = (const float*)d_in[2];
  const float* hw[3] = { (const float*)d_in[3],  (const float*)d_in[7],  (const float*)d_in[11] };
  const float* hb[3] = { (const float*)d_in[4],  (const float*)d_in[8],  (const float*)d_in[12] };
  const float* pw[3] = { (const float*)d_in[5],  (const float*)d_in[9],  (const float*)d_in[13] };
  const float* pb[3] = { (const float*)d_in[6],  (const float*)d_in[10], (const float*)d_in[14] };
  const float* W1 = (const float*)d_in[15];
  const float* b1 = (const float*)d_in[16];
  const float* W2 = (const float*)d_in[17];
  const float* b2 = (const float*)d_in[18];
  float* out = (float*)d_out;

  char* wp = (char*)d_ws;
  auto alloc = [&](size_t bytes) { char* p = wp; wp += (bytes + 255) & ~(size_t)255; return p; };
  f16*   sentF16 = (f16*)  alloc((size_t)2 * BATCH * SEQ * CPAD * 2);
  float* sentF32 = (float*)alloc((size_t)2 * BATCH * SEQ * CDIM * 4);
  f16*   Apad    = (f16*)  alloc((size_t)MPAD * CPAD * 6 * 2);
  f16*   W1T     = (f16*)  alloc((size_t)NPAD2 * KPADF * 2);
  float* convOut = (float*)alloc((size_t)MPAD * BATCH * SEQ * 4);
  float* apool   = (float*)alloc((size_t)2 * 3 * 3 * BATCH * NHOL * 4);
  float* infp    = (float*)alloc((size_t)2 * 3 * BATCH * 4);
  float* bpool   = (float*)alloc((size_t)2 * 2 * 3 * BATCH * KPF * CDIM * 4);
  f16*   featF16 = (f16*)  alloc((size_t)BATCH * KPADF * 2);
  float* h32     = (float*)alloc((size_t)BATCH * NPAD2 * 4);

  k_gather<<<dim3(SEQ, BATCH, 2), 320, 0, stream>>>(sa, sb, emb, sentF16, sentF32);
  {
    int total = MPAD * CPAD * 6;
    k_prepA<<<(total + 255) / 256, 256, 0, stream>>>(hw[0], hw[1], hw[2], Apad);
  }
  {
    long total = (long)NPAD2 * KPADF;
    k_prepW1<<<(int)((total + 255) / 256), 256, 0, stream>>>(W1, W1T);
  }

  const f16* Abase[3] = { Apad, Apad + MPAD * CPAD, Apad + (size_t)MPAD * CPAD * 3 };
  for (int s = 0; s < 2; ++s) {
    const f16* sentS = sentF16 + (size_t)s * BATCH * SEQ * CPAD;
    for (int wi = 0; wi < 3; ++wi) {
      int Lo = SEQ - wi;
      int tilesN = (BATCH * Lo) / 16;                 // 512 / 504 / 496, all % 8 == 0
      k_conv_gemm<<<dim3(tilesN / 8, MPAD / 16), 256, 0, stream>>>(
          Abase[wi], sentS, convOut, wi + 1, Lo);
      int n = BATCH * NHOL;
      k_pool_h<<<(n + 255) / 256, 256, 0, stream>>>(convOut, hb[wi], apool, s, wi, Lo);
    }
  }
  k_pool_inf<<<2 * BATCH, 256, 0, stream>>>(sentF32, infp);
  {
    int total = 2 * BATCH * CDIM * KPF;
    k_perdim<<<(total + 255) / 256, 256, 0, stream>>>(sentF32, pw[0], pb[0], pw[1], pb[1],
                                                      pw[2], pb[2], bpool);
  }
  k_features<<<BATCH, 256, 0, stream>>>(apool, infp, bpool, featF16);
  k_mlp_gemm<<<10, 256, 0, stream>>>(featF16, W1T, h32);
  k_head<<<1, 128, 0, stream>>>(h32, b1, W2, b2, out);
}
